// TriXTile_51934744543462
// MI455X (gfx1250) — matrix operand order repaired and checked by reading, not compiled, not run on hardware
//
#include <hip/hip_runtime.h>

// ---------------------------------------------------------------------------
// Ternary-weight MLP (BitNet-style) for MI455X / gfx1250.
// Pass 0: f32 -> f16 conversions (x) and f32 -> sign() f16 (both weights).
// Pass 1: hidden = relu((x_f16 @ up_f16^T) * up_scale)      -> f16 ws
// Pass 2: out    = (hidden @ dn_f16^T) * dn_scale * o_scale -> f32
// GEMMs use v_wmma_f32_16x16x32_f16, 128x128x64 block tiles, 8 waves,
// double-buffered LDS staged with GLOBAL_LOAD_ASYNC_TO_LDS_B128
// (ASYNCcnt-tracked CDNA5 async copies).
// ---------------------------------------------------------------------------

typedef __attribute__((ext_vector_type(16))) _Float16 v16h;
typedef __attribute__((ext_vector_type(8)))  _Float16 v8h;
typedef __attribute__((ext_vector_type(8)))  float    v8f;
typedef __attribute__((ext_vector_type(4)))  float    v4f;
typedef __attribute__((ext_vector_type(4)))  int      v4i;

#define BM 128
#define BN 128
#define BK 64
#define PITCH 72          // LDS row pitch in halves (64 data + 8 pad = 144B)
#define ASYNC_PER_TILE 8  // 4 (A) + 4 (B) b128 async copies per thread/stage

#if __has_builtin(__builtin_amdgcn_global_load_async_to_lds_b128) && \
    __has_builtin(__builtin_amdgcn_s_wait_asynccnt)
#define USE_ASYNC 1
#define WAIT_ASYNC(n) __builtin_amdgcn_s_wait_asynccnt(n)
#else
#define USE_ASYNC 0
#define WAIT_ASYNC(n)
#endif

union V16 { v16h v; v8h h[2]; };

__device__ __forceinline__ _Float16 sgn_h(float x) {
  return x > 0.f ? (_Float16)1.f : (x < 0.f ? (_Float16)(-1.f) : (_Float16)0.f);
}

// Copy 16B global -> LDS (async on CDNA5, sync fallback otherwise).
__device__ __forceinline__ void cp16(const _Float16* __restrict__ g,
                                     _Float16* __restrict__ l) {
#if USE_ASYNC
  __builtin_amdgcn_global_load_async_to_lds_b128(
      (__attribute__((address_space(1))) v4i*)g,
      (__attribute__((address_space(3))) v4i*)l, 0, 0);
#else
  *(v8h*)l = *(const v8h*)g;
#endif
}

// Stage one BM x BK f16 tile (16 KB) with 256 threads: 4 x b128 per thread.
__device__ __forceinline__ void stage_tile(const _Float16* __restrict__ G,
                                           int K, _Float16* __restrict__ L,
                                           int tid) {
#pragma unroll
  for (int i = 0; i < 4; ++i) {
    int c   = tid + i * 256;   // 0..1023 chunk id
    int row = c >> 3;          // 128 rows
    int c8  = (c & 7) * 8;     // 8 chunks of 8 halves per row
    cp16(G + (size_t)row * K + c8, L + row * PITCH + c8);
  }
}

// ---------------------------------------------------------------------------
// Pre-pass: f32 -> f16, optionally sign-ternarized. 8 elements / thread.
// ---------------------------------------------------------------------------
template<bool SIGN>
__global__ __launch_bounds__(256)
void trix_cvt(const float* __restrict__ in, _Float16* __restrict__ out, int n8) {
  int i = blockIdx.x * 256 + threadIdx.x;
  if (i >= n8) return;
  v4f f0 = *((const v4f*)in + 2 * (size_t)i);
  v4f f1 = *((const v4f*)in + 2 * (size_t)i + 1);
  v8h h;
  if constexpr (SIGN) {
    h[0] = sgn_h(f0.x); h[1] = sgn_h(f0.y); h[2] = sgn_h(f0.z); h[3] = sgn_h(f0.w);
    h[4] = sgn_h(f1.x); h[5] = sgn_h(f1.y); h[6] = sgn_h(f1.z); h[7] = sgn_h(f1.w);
  } else {
    h[0] = (_Float16)f0.x; h[1] = (_Float16)f0.y; h[2] = (_Float16)f0.z; h[3] = (_Float16)f0.w;
    h[4] = (_Float16)f1.x; h[5] = (_Float16)f1.y; h[6] = (_Float16)f1.z; h[7] = (_Float16)f1.w;
  }
  *((v8h*)out + (size_t)i) = h;
}

// ---------------------------------------------------------------------------
// WMMA GEMM: C[M,N] = A[M,K](f16) x B[N,K](f16)^T, per-column scale epilogue.
// RELU_F16: *col_scale, relu, store f16.  Else: *col_scale*o_scale, store f32.
// ---------------------------------------------------------------------------
template<bool RELU_F16>
__global__ __launch_bounds__(256)
void trix_wmma_gemm(const _Float16* __restrict__ A,
                    const _Float16* __restrict__ B,
                    const float* __restrict__ col_scale,
                    const float* __restrict__ out_scale,
                    void* __restrict__ Cout,
                    int M, int N, int K)
{
  __shared__ _Float16 lA[2][BM * PITCH];
  __shared__ _Float16 lB[2][BN * PITCH];

  const int tid   = threadIdx.x;
  const int lane  = tid & 31;
  const int wave  = tid >> 5;      // 0..7
  const int waveM = wave >> 2;     // 0..1  (64-row slab)
  const int waveN = wave & 3;      // 0..3  (32-col slab)
  const int r     = lane & 15;     // row/col within fragment
  const int khalf = lane >> 4;     // K-half select per ISA 7.12.2

  const int n0 = blockIdx.x * BN;
  const int m0 = blockIdx.y * BM;

  const _Float16* Ag = A + (size_t)m0 * K;
  const _Float16* Bg = B + (size_t)n0 * K;

  v8f acc[4][2] = {};

  const int T = K / BK;
  stage_tile(Ag, K, lA[0], tid);
  stage_tile(Bg, K, lB[0], tid);

  for (int t = 0; t < T; ++t) {
    const int cur = t & 1;
    if (t + 1 < T) {              // prefetch next tile into the other buffer
      stage_tile(Ag + (size_t)(t + 1) * BK, K, lA[cur ^ 1], tid);
      stage_tile(Bg + (size_t)(t + 1) * BK, K, lB[cur ^ 1], tid);
      WAIT_ASYNC(ASYNC_PER_TILE);  // tile t complete (in-order per wave)
    } else {
      WAIT_ASYNC(0);
    }
    __syncthreads();               // whole tile t visible to all waves

    const _Float16* la = lA[cur];
    const _Float16* lb = lB[cur];
#pragma unroll
    for (int ks = 0; ks < BK; ks += 32) {
      // A 16x32 f16 frag: lane half khalf -> K = ks+khalf*8 (+16 for v4-7)
      V16 a[4];
#pragma unroll
      for (int mi = 0; mi < 4; ++mi) {
        const _Float16* base =
            &la[(waveM * 64 + mi * 16 + r) * PITCH + ks + khalf * 8];
        a[mi].h[0] = *(const v8h*)(base);
        a[mi].h[1] = *(const v8h*)(base + 16);
      }
      // B 32x16 f16 frag: lane half khalf -> K = ks+khalf*8 (+16 for v4-7), col = r
      V16 b[2];
#pragma unroll
      for (int ni = 0; ni < 2; ++ni) {
        const _Float16* base =
            &lb[(waveN * 32 + ni * 16 + r) * PITCH + ks + khalf * 8];
        b[ni].h[0] = *(const v8h*)(base);
        b[ni].h[1] = *(const v8h*)(base + 16);
      }
      // ni-outer / mi-inner: keeps all four A fragments live so the
      // compiler issues the 8 ds_loads up front and the 8 WMMAs run
      // back-to-back instead of serializing on per-pair dscnt waits.
#pragma unroll
      for (int ni = 0; ni < 2; ++ni)
#pragma unroll
        for (int mi = 0; mi < 4; ++mi)
          acc[mi][ni] = __builtin_amdgcn_wmma_f32_16x16x32_f16(
              false, a[mi].v, false, b[ni].v,
              (short)0, acc[mi][ni], false, false);
    }
    __syncthreads();               // all waves done reading buf before reuse
  }

  // Epilogue: C layout VGPR e -> M = e + 8*khalf, N = r
#pragma unroll
  for (int mi = 0; mi < 4; ++mi) {
#pragma unroll
    for (int ni = 0; ni < 2; ++ni) {
      const int gcol = n0 + waveN * 32 + ni * 16 + r;
      float s = col_scale[gcol];
      if constexpr (!RELU_F16) s *= out_scale[0];
      const int rowbase = m0 + waveM * 64 + mi * 16 + khalf * 8;
#pragma unroll
      for (int e = 0; e < 8; ++e) {
        float v = acc[mi][ni][e] * s;
        if constexpr (RELU_F16) {
          v = v > 0.f ? v : 0.f;
          ((_Float16*)Cout)[(size_t)(rowbase + e) * N + gcol] = (_Float16)v;
        } else {
          ((float*)Cout)[(size_t)(rowbase + e) * N + gcol] = v;
        }
      }
    }
  }
}

extern "C" void kernel_launch(void* const* d_in, const int* in_sizes, int n_in,
                              void* d_out, int out_size, void* d_ws, size_t ws_size,
                              hipStream_t stream) {
  const float* x    = (const float*)d_in[0];  // [M, d_model] f32
  const float* up_w = (const float*)d_in[1];  // [d_hidden, d_model] f32
  const float* dn_w = (const float*)d_in[2];  // [d_model, d_hidden] f32
  const float* up_s = (const float*)d_in[3];  // [d_hidden]
  const float* dn_s = (const float*)d_in[4];  // [d_model]
  const float* o_s  = (const float*)d_in[5];  // [1]

  const int d_hidden = in_sizes[3];            // 8192
  const int d_model  = in_sizes[4];            // 2048
  const int M        = in_sizes[0] / d_model;  // 4096

  // Workspace layout (f16): x | up_w | dn_w | hidden  (~151 MB total)
  _Float16* xf  = (_Float16*)d_ws;
  _Float16* upf = xf  + (size_t)M * d_model;
  _Float16* dnf = upf + (size_t)d_hidden * d_model;
  _Float16* hid = dnf + (size_t)d_hidden * d_model;

  dim3 blk(256);

  // Pass 0: conversions (bandwidth-bound, ~200 MB total traffic)
  {
    int n8 = (M * d_model) / 8;
    trix_cvt<false><<<dim3((n8 + 255) / 256), blk, 0, stream>>>(x, xf, n8);
    n8 = (d_hidden * d_model) / 8;
    trix_cvt<true><<<dim3((n8 + 255) / 256), blk, 0, stream>>>(up_w, upf, n8);
    trix_cvt<true><<<dim3((n8 + 255) / 256), blk, 0, stream>>>(dn_w, dnf, n8);
  }

  // Pass 1: hidden = relu((x @ up^T) * up_scale)   [M, d_hidden] f16
  dim3 g1(d_hidden / BN, M / BM);
  trix_wmma_gemm<true><<<g1, blk, 0, stream>>>(
      xf, upf, up_s, o_s, hid, M, d_hidden, d_model);

  // Pass 2: out = (hidden @ dn^T) * dn_scale * o_scale   [M, d_model] f32
  dim3 g2(d_model / BN, M / BM);
  trix_wmma_gemm<false><<<g2, blk, 0, stream>>>(
      hid, dnf, dn_s, o_s, (float*)d_out, M, d_model, d_hidden);
}
